// SGFormerEdgeEmbs_14199161880982
// MI455X (gfx1250) — matrix operand order repaired and checked by reading, not compiled, not run on hardware
//
#include <hip/hip_runtime.h>

#define EPS_LN 1e-5f
#define ALPHA_MIX 0.5f

typedef __attribute__((ext_vector_type(16))) __bf16 v16bf;
typedef __attribute__((ext_vector_type(8)))  float  v8f;
typedef unsigned int u32x4 __attribute__((ext_vector_type(4)));
typedef int          i32x8 __attribute__((ext_vector_type(8)));
typedef int          i32x4 __attribute__((ext_vector_type(4)));

struct Pack32 { uint4 a, b; };

__device__ __forceinline__ unsigned f2bf(float x) {
    unsigned u = __float_as_uint(x);
    return (u + 0x7FFFu + ((u >> 16) & 1u)) >> 16;   // round-to-nearest-even
}
__device__ __forceinline__ unsigned pk2(float x, float y) {
    return f2bf(x) | (f2bf(y) << 16);
}
__device__ __forceinline__ v16bf as16(uint4 a, uint4 b) {
    Pack32 p{a, b};
    return __builtin_bit_cast(v16bf, p);
}

// ---------------------------------------------------------------------------
// GEMM: out[n,j] = sum_k A[n,k] * W[j,k] (+bias[j]) for n in [0,nrows), j,k in [0,256)
// W given as bf16 bits, row-major [256][256].
// MODE 0: +bias, LayerNorm, ReLU            (fc layer)
// MODE 1: +bias, atomic sum of squares      (q)
// MODE 2: +bias, sumsq + column sums        (k)
// MODE 3: +bias                             (v)
// MODE 4: plain                             (q @ S)
// Block: 256 threads = 8 waves; wave computes 16 rows x 256 cols (16 WMMA tiles).
// Weight k-slices are DMA'd into LDS by the Tensor Data Mover (TDM) with
// hardware padding producing the 72-ushort bank-conflict-free pitch.
// ---------------------------------------------------------------------------
template <int MODE>
__global__ void gemm256(const float* __restrict__ A,
                        const unsigned short* __restrict__ W,
                        const float* __restrict__ bias,
                        float* __restrict__ out, int nrows,
                        const float* __restrict__ lng, const float* __restrict__ lnb,
                        float* __restrict__ sumsq, float* __restrict__ colsum) {
    constexpr int PITCH = 72;                       // 64 + 8 pad (bank-conflict-free)
    __shared__ unsigned short wl[256 * PITCH];      // 36 KB k-slice of W
    const int tid  = threadIdx.x;
    const int wave = tid >> 5, lane = tid & 31;
    const int ll = lane & 15, half = lane >> 4;
    const int row0 = blockIdx.x * 128 + wave * 16;

    v8f acc[16];
#pragma unroll
    for (int t = 0; t < 16; ++t)
#pragma unroll
        for (int e = 0; e < 8; ++e) acc[t][e] = 0.f;

    const int mrow = row0 + ll;
    const int mclamp = (mrow < nrows) ? mrow : (nrows - 1);
    const float* ap_base = A + (size_t)mclamp * 256;

    for (int ks = 0; ks < 256; ks += 64) {
        __syncthreads();   // previous slice fully consumed
#if __has_builtin(__builtin_amdgcn_tensor_load_to_lds)
        {
            // TDM: DMA W[0:256, ks:ks+64] (bf16) into LDS with 16B pad per row.
            const unsigned lds_off = (unsigned)(size_t)(void*)wl;  // LDS aperture: low 32b = offset
            const unsigned long long ga =
                (unsigned long long)(size_t)(const void*)W + (unsigned)(ks * 2);
            u32x4 g0;
            g0[0] = 1u;                                   // count=1, user mode, no gather
            g0[1] = lds_off;                              // lds_addr
            g0[2] = (unsigned)ga;                         // global_addr[31:0]
            g0[3] = (unsigned)((ga >> 32) & 0x01FFFFFFu)  // global_addr[56:32]
                    | (2u << 30);                         // type = 2 ("image")
            i32x8 g1;
            g1[0] = (int)((1u << 16)      // data_size = 2 bytes
                          | (1u << 20)    // pad_enable
                          | (4u << 22)    // pad_interval: 32 DWORDs (=64 elems)
                          | (3u << 25));  // pad_amount: 4 DWORDs (=8 elems)
            g1[1] = (int)(256u << 16);    // tensor_dim0 = 256 (low16 @ bits63:48)
            g1[2] = (int)(256u << 16);    // tensor_dim0 hi=0 | tensor_dim1 = 256 (low16)
            g1[3] = (int)(64u << 16);     // tensor_dim1 hi=0 | tile_dim0 = 64
            g1[4] = 256;                  // tile_dim1 = 256, tile_dim2 = 0
            g1[5] = 256;                  // tensor_dim0_stride = 256 (low32)
            g1[6] = 0;                    // stride hi | tensor_dim1_stride low
            g1[7] = 0;
            i32x4 gz; gz[0] = 0; gz[1] = 0; gz[2] = 0; gz[3] = 0;
            i32x8 gz8;
#pragma unroll
            for (int i = 0; i < 8; ++i) gz8[i] = 0;
            if (tid < 32) {               // one wave issues the DMA
                __builtin_amdgcn_tensor_load_to_lds(g0, g1, gz, gz, gz8, 0);
                __builtin_amdgcn_s_wait_tensorcnt(0);
            }
        }
#else
        {   // fallback: cooperative stage of W[:, ks:ks+64] -> LDS (one row per thread)
            const unsigned short* src = W + tid * 256 + ks;
            unsigned short* dst = &wl[tid * PITCH];
#pragma unroll
            for (int i = 0; i < 8; ++i)
                *(uint4*)(dst + 8 * i) = *(const uint4*)(src + 8 * i);
        }
#endif
        __syncthreads();   // slice visible to all waves
        if (ks < 192) __builtin_prefetch(ap_base + ks + 64, 0, 3);
#pragma unroll
        for (int kk = 0; kk < 64; kk += 32) {
            // A operand: lane ll holds row, K interleave per CDNA5 16-bit A layout
            const float* ap = ap_base + (ks + kk) + half * 8;
            float4 f0 = *(const float4*)(ap);
            float4 f1 = *(const float4*)(ap + 4);
            float4 f2 = *(const float4*)(ap + 16);
            float4 f3 = *(const float4*)(ap + 20);
            uint4 al = make_uint4(pk2(f0.x, f0.y), pk2(f0.z, f0.w),
                                  pk2(f1.x, f1.y), pk2(f1.z, f1.w));
            uint4 ah = make_uint4(pk2(f2.x, f2.y), pk2(f2.z, f2.w),
                                  pk2(f3.x, f3.y), pk2(f3.z, f3.w));
            v16bf a = as16(al, ah);
#pragma unroll
            for (int t = 0; t < 16; ++t) {
                const unsigned short* bp = &wl[(t * 16 + ll) * PITCH + kk + half * 8];
                v16bf b = as16(*(const uint4*)bp, *(const uint4*)(bp + 16));
                acc[t] = __builtin_amdgcn_wmma_f32_16x16x32_bf16(
                    false, a, false, b, (short)0, acc[t], false, false);
            }
        }
    }

    if constexpr (MODE != 4) {
#pragma unroll
        for (int t = 0; t < 16; ++t) {
            float bj = bias[t * 16 + ll];
#pragma unroll
            for (int e = 0; e < 8; ++e) acc[t][e] += bj;
        }
    }

    if constexpr (MODE == 0) {
        // fused LayerNorm + ReLU; row m = row0 + e + 8*half lives in 16 lanes of a half
#pragma unroll
        for (int e = 0; e < 8; ++e) {
            float s = 0.f, ss = 0.f;
#pragma unroll
            for (int t = 0; t < 16; ++t) { float vv = acc[t][e]; s += vv; ss += vv * vv; }
#pragma unroll
            for (int d = 1; d < 16; d <<= 1) {
                s += __shfl_xor(s, d, 16);
                ss += __shfl_xor(ss, d, 16);
            }
            float mean = s * (1.f / 256.f);
            float var  = ss * (1.f / 256.f) - mean * mean;
            float inv  = rsqrtf(var + EPS_LN);
            int m = row0 + e + half * 8;
            if (m < nrows) {
                float* op = out + (size_t)m * 256;
#pragma unroll
                for (int t = 0; t < 16; ++t) {
                    int j = t * 16 + ll;
                    float val = (acc[t][e] - mean) * inv * lng[j] + lnb[j];
                    op[j] = fmaxf(val, 0.f);
                }
            }
        }
    } else {
#pragma unroll
        for (int e = 0; e < 8; ++e) {
            int m = row0 + e + half * 8;
            if (m < nrows) {
                float* op = out + (size_t)m * 256;
#pragma unroll
                for (int t = 0; t < 16; ++t) op[t * 16 + ll] = acc[t][e];
            }
        }
        if constexpr (MODE == 1 || MODE == 2) {
            float ssq = 0.f;
#pragma unroll
            for (int e = 0; e < 8; ++e) {
                int m = row0 + e + half * 8;
                float msk = (m < nrows) ? 1.f : 0.f;
#pragma unroll
                for (int t = 0; t < 16; ++t) { float vv = acc[t][e] * msk; ssq += vv * vv; }
            }
#pragma unroll
            for (int d = 1; d < 32; d <<= 1) ssq += __shfl_xor(ssq, d, 32);
            if (lane == 0) atomicAdd(sumsq, ssq);
        }
        if constexpr (MODE == 2) {
#pragma unroll
            for (int t = 0; t < 16; ++t) {
                float cs = 0.f;
#pragma unroll
                for (int e = 0; e < 8; ++e) {
                    int m = row0 + e + half * 8;
                    cs += (m < nrows) ? acc[t][e] : 0.f;
                }
                cs += __shfl_xor(cs, 16, 32);
                if (half == 0) atomicAdd(&colsum[t * 16 + ll], cs);
            }
        }
    }
}

// ---------------------------------------------------------------------------
// S += k^T v  (S: [256 feat][256 d], reduction over nrows)
// grid = (nsplit, 2): blockIdx.y selects 128 feats of k; each block loops its
// n-range in chunks of 32, staging k/v TRANSPOSED as bf16 in LDS so both WMMA
// operands are contiguous ds_load_b128 reads. fp32 atomics into S.
// ---------------------------------------------------------------------------
__global__ void __launch_bounds__(256)
ktv_kernel(const float* __restrict__ K, const float* __restrict__ V,
           float* __restrict__ S, int nrows, int rows_per_block) {
    constexpr int PIT = 40;                          // 32 + 8 pad
    __shared__ unsigned short kt[128 * PIT];         // k^T tile  [feat][n]
    __shared__ unsigned short vt[256 * PIT];         // v^T tile  [d][n]
    const int tid  = threadIdx.x;
    const int wave = tid >> 5, lane = tid & 31;
    const int ll = lane & 15, half = lane >> 4;
    const int fbase = blockIdx.y * 128;
    const int n0 = blockIdx.x * rows_per_block;
    int nend = n0 + rows_per_block;
    if (nend > nrows) nend = nrows;

    v8f acc[16];
#pragma unroll
    for (int t = 0; t < 16; ++t)
#pragma unroll
        for (int e = 0; e < 8; ++e) acc[t][e] = 0.f;

    for (int nb = n0; nb < nend; nb += 32) {
        __syncthreads();
        {
            int r = tid >> 3;               // 0..31 rows of the chunk
            int gr = nb + r;
            bool ok = gr < nrows;
            int gc = ok ? gr : 0;
            // k^T stage: 32 x 128 (cols fbase..fbase+127)
            int c0 = (tid & 7) * 16;
            const float* kp = K + (size_t)gc * 256 + fbase + c0;
#pragma unroll
            for (int i = 0; i < 4; ++i) {
                float4 f = *(const float4*)(kp + 4 * i);
                if (!ok) f = make_float4(0.f, 0.f, 0.f, 0.f);
                kt[(c0 + 4 * i + 0) * PIT + r] = (unsigned short)f2bf(f.x);
                kt[(c0 + 4 * i + 1) * PIT + r] = (unsigned short)f2bf(f.y);
                kt[(c0 + 4 * i + 2) * PIT + r] = (unsigned short)f2bf(f.z);
                kt[(c0 + 4 * i + 3) * PIT + r] = (unsigned short)f2bf(f.w);
            }
            // v^T stage: 32 x 256
            int c1 = (tid & 7) * 32;
            const float* vp = V + (size_t)gc * 256 + c1;
#pragma unroll
            for (int i = 0; i < 8; ++i) {
                float4 f = *(const float4*)(vp + 4 * i);
                if (!ok) f = make_float4(0.f, 0.f, 0.f, 0.f);
                vt[(c1 + 4 * i + 0) * PIT + r] = (unsigned short)f2bf(f.x);
                vt[(c1 + 4 * i + 1) * PIT + r] = (unsigned short)f2bf(f.y);
                vt[(c1 + 4 * i + 2) * PIT + r] = (unsigned short)f2bf(f.z);
                vt[(c1 + 4 * i + 3) * PIT + r] = (unsigned short)f2bf(f.w);
            }
        }
        __syncthreads();
        const unsigned short* apb = &kt[(wave * 16 + ll) * PIT];
        v16bf a = as16(*(const uint4*)(apb + half * 8),
                       *(const uint4*)(apb + 16 + half * 8));
#pragma unroll
        for (int t = 0; t < 16; ++t) {
            const unsigned short* bp = &vt[(t * 16 + ll) * PIT + half * 8];
            v16bf b = as16(*(const uint4*)bp, *(const uint4*)(bp + 16));
            acc[t] = __builtin_amdgcn_wmma_f32_16x16x32_bf16(
                false, a, false, b, (short)0, acc[t], false, false);
        }
    }

#pragma unroll
    for (int t = 0; t < 16; ++t)
#pragma unroll
        for (int e = 0; e < 8; ++e) {
            int mfeat = fbase + wave * 16 + e + half * 8;
            int d = t * 16 + ll;
            atomicAdd(&S[mfeat * 256 + d], acc[t][e]);
        }
}

// S_scaled (transposed into B-operand weight layout, bf16) and ksum_scaled
__global__ void scale_kernel(const float* __restrict__ Sacc,
                             const float* __restrict__ sq, const float* __restrict__ sk,
                             const float* __restrict__ ksum,
                             unsigned short* __restrict__ wS,
                             float* __restrict__ ksum_s) {
    float inv = rsqrtf(fmaxf(sq[0] * sk[0], 1e-30f));
    for (int e = blockIdx.x * 256 + threadIdx.x; e < 65536; e += gridDim.x * 256) {
        int m = e >> 8, d = e & 255;
        wS[d * 256 + m] = (unsigned short)f2bf(Sacc[e] * inv);
    }
    if (blockIdx.x == 0) ksum_s[threadIdx.x] = ksum[threadIdx.x] * inv;
}

__device__ __forceinline__ void ld8(const float* p, float* o) {
    float4 a = *(const float4*)p;
    float4 b = *(const float4*)(p + 4);
    o[0] = a.x; o[1] = a.y; o[2] = a.z; o[3] = a.w;
    o[4] = b.x; o[5] = b.y; o[6] = b.z; o[7] = b.w;
}

// attn = (num + N*v)/(q.ksum_s + N); h = relu(LN(0.5*attn + 0.5*prev))
__global__ void attn_ln_kernel(const float* __restrict__ q, const float* __restrict__ num,
                               const float* __restrict__ v, const float* __restrict__ prev,
                               const float* __restrict__ ksum_s,
                               const float* __restrict__ lng, const float* __restrict__ lnb,
                               float* __restrict__ out, int nrows) {
    const int wave = threadIdx.x >> 5, lane = threadIdx.x & 31;
    const int n = blockIdx.x * 8 + wave;
    if (n >= nrows) return;
    const int j0 = lane * 8;
    const size_t base = (size_t)n * 256 + j0;

    float qv[8], sv[8];
    ld8(q + base, qv); ld8(ksum_s + j0, sv);
    float qd = 0.f;
#pragma unroll
    for (int i = 0; i < 8; ++i) qd += qv[i] * sv[i];
#pragma unroll
    for (int d = 1; d < 32; d <<= 1) qd += __shfl_xor(qd, d, 32);
    const float fN = (float)nrows;
    const float rden = 1.f / (qd + fN);

    float nm[8], vv[8], pv[8], val[8];
    ld8(num + base, nm); ld8(v + base, vv); ld8(prev + base, pv);
    float s = 0.f, ss = 0.f;
#pragma unroll
    for (int i = 0; i < 8; ++i) {
        float a = (nm[i] + fN * vv[i]) * rden;
        val[i] = ALPHA_MIX * a + (1.f - ALPHA_MIX) * pv[i];
        s += val[i]; ss += val[i] * val[i];
    }
#pragma unroll
    for (int d = 1; d < 32; d <<= 1) { s += __shfl_xor(s, d, 32); ss += __shfl_xor(ss, d, 32); }
    float mean = s * (1.f / 256.f);
    float var  = ss * (1.f / 256.f) - mean * mean;
    float inv  = rsqrtf(var + EPS_LN);

    float gv[8], bv[8];
    ld8(lng + j0, gv); ld8(lnb + j0, bv);
    float r[8];
#pragma unroll
    for (int i = 0; i < 8; ++i) r[i] = fmaxf((val[i] - mean) * inv * gv[i] + bv[i], 0.f);
    *(float4*)(out + base)     = make_float4(r[0], r[1], r[2], r[3]);
    *(float4*)(out + base + 4) = make_float4(r[4], r[5], r[6], r[7]);
}

__global__ void cvt_kernel(const float* __restrict__ src, unsigned short* __restrict__ dst, int n) {
    for (int i = blockIdx.x * blockDim.x + threadIdx.x; i < n; i += gridDim.x * blockDim.x)
        dst[i] = (unsigned short)f2bf(src[i]);
}
__global__ void zero_kernel(float* __restrict__ p, int n) {
    for (int i = blockIdx.x * blockDim.x + threadIdx.x; i < n; i += gridDim.x * blockDim.x)
        p[i] = 0.f;
}

// ---------------------------------------------------------------------------
extern "C" void kernel_launch(void* const* d_in, const int* in_sizes, int n_in,
                              void* d_out, int out_size, void* d_ws, size_t ws_size,
                              hipStream_t stream) {
    (void)n_in; (void)out_size; (void)ws_size;
    const float* x    = (const float*)d_in[0];
    const float* fc_w = (const float*)d_in[1];
    const float* fc_b = (const float*)d_in[2];
    const float* Wq_w = (const float*)d_in[3];
    const float* Wq_b = (const float*)d_in[4];
    const float* Wk_w = (const float*)d_in[5];
    const float* Wk_b = (const float*)d_in[6];
    const float* Wv_w = (const float*)d_in[7];
    const float* Wv_b = (const float*)d_in[8];
    const float* ln_g = (const float*)d_in[9];
    const float* ln_b = (const float*)d_in[10];
    float* out = (float*)d_out;
    const int nrows = in_sizes[0] / 256;
    const size_t B = (size_t)nrows * 256;

    float* hA = (float*)d_ws;
    float* hB = hA + B;
    float* qb = hB + B;
    float* kb = qb + B;          // k; reused as numraw after ktv
    float* vb = kb + B;
    unsigned short* wb = (unsigned short*)(vb + B);   // 7 bf16 weight slots + wS
    unsigned short* wS = wb + 65536 * 7;
    float* Sacc   = (float*)(wb + 65536 * 8);
    float* ksum   = Sacc + 65536;
    float* sq     = Sacc + 65792;
    float* sk     = Sacc + 65793;
    float* ksum_s = Sacc + 65800;

    cvt_kernel<<<64, 256, 0, stream>>>(fc_w, wb, 65536);
    for (int l = 0; l < 2; ++l) {
        cvt_kernel<<<64, 256, 0, stream>>>(Wq_w + l * 65536, wb + 65536 * (1 + 3 * l + 0), 65536);
        cvt_kernel<<<64, 256, 0, stream>>>(Wk_w + l * 65536, wb + 65536 * (1 + 3 * l + 1), 65536);
        cvt_kernel<<<64, 256, 0, stream>>>(Wv_w + l * 65536, wb + 65536 * (1 + 3 * l + 2), 65536);
    }

    dim3 gblk((nrows + 127) / 128);
    gemm256<0><<<gblk, 256, 0, stream>>>(x, wb, fc_b, hA, nrows, ln_g, ln_b, nullptr, nullptr);

    const float* h = hA;
    for (int l = 0; l < 2; ++l) {
        zero_kernel<<<64, 256, 0, stream>>>(Sacc, 65794);
        gemm256<1><<<gblk, 256, 0, stream>>>(h, wb + 65536 * (1 + 3 * l + 0), Wq_b + l * 256,
                                             qb, nrows, nullptr, nullptr, sq, nullptr);
        gemm256<2><<<gblk, 256, 0, stream>>>(h, wb + 65536 * (1 + 3 * l + 1), Wk_b + l * 256,
                                             kb, nrows, nullptr, nullptr, sk, ksum);
        gemm256<3><<<gblk, 256, 0, stream>>>(h, wb + 65536 * (1 + 3 * l + 2), Wv_b + l * 256,
                                             vb, nrows, nullptr, nullptr, nullptr, nullptr);
        const int nsplit = 64;
        int rpb = ((nrows + nsplit * 32 - 1) / (nsplit * 32)) * 32;
        ktv_kernel<<<dim3(nsplit, 2), 256, 0, stream>>>(kb, vb, Sacc, nrows, rpb);
        scale_kernel<<<64, 256, 0, stream>>>(Sacc, sq, sk, ksum, wS, ksum_s);
        gemm256<4><<<gblk, 256, 0, stream>>>(qb, wS, nullptr, kb, nrows,
                                             nullptr, nullptr, nullptr, nullptr);
        float* hn = (l == 0) ? hB : out;
        attn_ln_kernel<<<(nrows + 7) / 8, 256, 0, stream>>>(
            qb, kb, vb, h, ksum_s, ln_g + (l + 1) * 256, ln_b + (l + 1) * 256, hn, nrows);
        h = hn;
    }
}
